// MoEResBlock_47485158425014
// MI455X (gfx1250) — matrix operand order repaired and checked by reading, not compiled, not run on hardware
//
#include <hip/hip_runtime.h>

typedef __bf16 bf16;
typedef __attribute__((ext_vector_type(16))) __bf16 v16bf;
typedef __attribute__((ext_vector_type(8)))  __bf16 v8bf;
typedef __attribute__((ext_vector_type(8)))  float  v8f;

#define NPIX     131072L            // B*H*W = 32*64*64
#define NELEM    8388608L           // NPIX*64
#define PADELEM  8921088L           // 32*66*66*64 (zero-padded NHWC)
#define LWSTRIDE 72                 // LDS weight row stride in halves (16B aligned, conflict-free)

static __device__ __forceinline__ v16bf ldfrag(const bf16* p) {
    v8bf lo = *(const v8bf*)p;
    v8bf hi = *(const v8bf*)(p + 8);
    v16bf r;
    #pragma unroll
    for (int i = 0; i < 8; ++i) { r[i] = lo[i]; r[8 + i] = hi[i]; }
    return r;
}

// ---------------- prep: x (NCHW f32) -> padded NHWC bf16, LDS-tiled transpose ----------------
__global__ __launch_bounds__(256) void k_x2nhwc(const float* __restrict__ x, bf16* __restrict__ xp) {
    __shared__ float tile[64][65];
    const int t   = threadIdx.x;
    const int b   = blockIdx.x >> 6;
    const int row = blockIdx.x & 63;                       // h
    const float* src = x + (long)b * 64 * 4096 + row * 64; // + c*4096 + w
    for (int i = t; i < 4096; i += 256) {                  // coalesced: w inner
        int c = i >> 6, w = i & 63;
        tile[c][w] = src[(long)c * 4096 + w];
    }
    __syncthreads();
    bf16* dst = xp + (((long)b * 66 + row + 1) * 66 + 1) * 64;  // + w*64 + c
    for (int i = t; i < 4096; i += 256) {                  // coalesced: c inner
        int w = i >> 6, c = i & 63;
        dst[(long)w * 64 + c] = (bf16)tile[c][w];
    }
}

// ---------------- prep: conv weights OIHW f32 -> [e][tap*64+ci][co] bf16 ----------------
__global__ __launch_bounds__(256) void k_wprep(const float* __restrict__ w1, const float* __restrict__ w2,
                                               bf16* __restrict__ wt1, bf16* __restrict__ wt2) {
    long i = (long)blockIdx.x * 256 + threadIdx.x;
    if (i >= 2L * 294912L) return;
    const float* src = (i < 294912L) ? w1 : w2;
    bf16* dst = (i < 294912L) ? wt1 : wt2;
    long r = i % 294912L;
    int e  = (int)(r / 36864L);
    int q  = (int)(r % 36864L);
    int k  = q >> 6, co = q & 63;
    int tap = k >> 6, ci = k & 63;
    long inIdx = ((((long)e * 64 + co) * 64 + ci) * 9) + tap;
    dst[r] = (bf16)src[inIdx];
}

// ---------------- router: pooled means ----------------
__global__ __launch_bounds__(256) void k_pool(const float* __restrict__ x, float* __restrict__ pooled) {
    __shared__ float ss[256];
    int t = threadIdx.x;
    const float* p = x + (long)blockIdx.x * 4096;
    float s = 0.f;
    for (int i = t; i < 4096; i += 256) s += p[i];
    ss[t] = s; __syncthreads();
    for (int off = 128; off > 0; off >>= 1) {
        if (t < off) ss[t] += ss[t + off];
        __syncthreads();
    }
    if (t == 0) pooled[blockIdx.x] = ss[0] / 4096.f;
}

// ---------------- router: logits/softmax/top2/gates/aux ----------------
__global__ __launch_bounds__(256) void k_router(const float* __restrict__ pooled, const float* __restrict__ rw,
                                                float* __restrict__ gates, float* __restrict__ aux_out) {
    __shared__ float probs[32][8];
    int t = threadIdx.x;
    {
        int b = t >> 3, e = t & 7;
        float s = 0.f;
        for (int c = 0; c < 64; ++c) s += pooled[b * 64 + c] * rw[e * 64 + c];
        probs[b][e] = s;
    }
    __syncthreads();
    if (t < 32) {
        float mx = probs[t][0];
        for (int e = 1; e < 8; ++e) mx = fmaxf(mx, probs[t][e]);
        float pr[8]; float sum = 0.f;
        for (int e = 0; e < 8; ++e) { pr[e] = expf(probs[t][e] - mx); sum += pr[e]; }
        for (int e = 0; e < 8; ++e) { pr[e] /= sum; probs[t][e] = pr[e]; }
        int i1 = 0;
        for (int e = 1; e < 8; ++e) if (pr[e] > pr[i1]) i1 = e;
        int i2 = (i1 == 0) ? 1 : 0;
        for (int e = 0; e < 8; ++e) if (e != i1 && pr[e] > pr[i2]) i2 = e;
        float wsum = pr[i1] + pr[i2];
        for (int e = 0; e < 8; ++e) gates[t * 8 + e] = 0.f;
        gates[t * 8 + i1] = pr[i1] / wsum;
        gates[t * 8 + i2] = pr[i2] / wsum;
    }
    __syncthreads();
    if (t == 0) {
        float aux = 0.f;
        for (int e = 0; e < 8; ++e) {
            float a = 0.f;
            for (int b = 0; b < 32; ++b) a += probs[b][e];
            a /= 32.f;
            aux += a * a;
        }
        aux_out[0] = 8.f * aux;
    }
}

// ---------------- 3x3 conv as implicit GEMM, bf16 WMMA, weights in LDS ----------------
// Each wave: 32M x 64N tile (2 A-frags x 4 N-tiles = 8 accumulators).
// Per K-chunk: 4 global b128 (A) + 64 ds 16-bit loads (B, 4 independent buffers) -> 8 back-to-back WMMAs.
// xin: padded NHWC bf16 [32][66][66][64]; wt: [576][64] bf16; out: NHWC f32 [NPIX][64]
__global__ __launch_bounds__(256) void k_conv(const bf16* __restrict__ xin, const bf16* __restrict__ wt,
                                              float* __restrict__ out) {
    extern __shared__ bf16 lw[];                 // [576][LWSTRIDE]
    const int t = threadIdx.x;
    for (int i = t; i < 4608; i += 256) {        // 576 rows * 8 segments of v8bf
        int row = i >> 3, seg = i & 7;
        *(v8bf*)(lw + row * LWSTRIDE + seg * 8) = *(const v8bf*)(wt + row * 64 + seg * 8);
    }
    __syncthreads();

    const int lane = t & 31;
    const int wid  = t >> 5;
    const int m0   = blockIdx.x * 256 + wid * 32;   // wave's 32 pixels (one half image row)
    const int b    = m0 >> 12;
    const int h    = (m0 >> 6) & 63;
    const int w0   = m0 & 63;                       // 0 or 32
    const int lm   = lane & 15;
    const int lhi  = lane >> 4;

    const bf16* pin0 = xin + (((long)b * 66 + h) * 66 + (w0 + lm)) * 64;
    const bf16* pin1 = pin0 + 16 * 64;              // pixels m0+16 .. m0+31

    v8f acc[2][4];
    #pragma unroll
    for (int ma = 0; ma < 2; ++ma)
        #pragma unroll
        for (int n = 0; n < 4; ++n)
            #pragma unroll
            for (int r = 0; r < 8; ++r) acc[ma][n][r] = 0.f;

    #pragma unroll
    for (int tap = 0; tap < 9; ++tap) {
        const int toff = ((tap / 3) * 66 + (tap % 3)) * 64;   // compile-time after unroll
        #pragma unroll
        for (int ch = 0; ch < 2; ++ch) {
            const int cb = ch * 32;
            // A fragments: lanes 0-15 hold K 0-15, lanes 16-31 hold K 16-31
            v16bf a0 = ldfrag(pin0 + toff + cb + lhi * 16);
            v16bf a1 = ldfrag(pin1 + toff + cb + lhi * 16);
            // All four B fragments preloaded into distinct registers (one ds clause, one wait)
            const bf16* wrow = lw + (tap * 64 + cb + lhi * 16) * LWSTRIDE + lm;
            v16bf bf0, bf1, bf2, bf3;
            #pragma unroll
            for (int s = 0; s < 16; ++s) {
                bf0[s] = wrow[s * LWSTRIDE];
                bf1[s] = wrow[s * LWSTRIDE + 16];
                bf2[s] = wrow[s * LWSTRIDE + 32];
                bf3[s] = wrow[s * LWSTRIDE + 48];
            }
            acc[0][0] = __builtin_amdgcn_wmma_f32_16x16x32_bf16(false, a0, false, bf0, (short)0, acc[0][0], false, false);
            acc[1][0] = __builtin_amdgcn_wmma_f32_16x16x32_bf16(false, a1, false, bf0, (short)0, acc[1][0], false, false);
            acc[0][1] = __builtin_amdgcn_wmma_f32_16x16x32_bf16(false, a0, false, bf1, (short)0, acc[0][1], false, false);
            acc[1][1] = __builtin_amdgcn_wmma_f32_16x16x32_bf16(false, a1, false, bf1, (short)0, acc[1][1], false, false);
            acc[0][2] = __builtin_amdgcn_wmma_f32_16x16x32_bf16(false, a0, false, bf2, (short)0, acc[0][2], false, false);
            acc[1][2] = __builtin_amdgcn_wmma_f32_16x16x32_bf16(false, a1, false, bf2, (short)0, acc[1][2], false, false);
            acc[0][3] = __builtin_amdgcn_wmma_f32_16x16x32_bf16(false, a0, false, bf3, (short)0, acc[0][3], false, false);
            acc[1][3] = __builtin_amdgcn_wmma_f32_16x16x32_bf16(false, a1, false, bf3, (short)0, acc[1][3], false, false);
        }
    }
    // D layout: row M = r + 8*lhi, col N = n*16 + lm
    #pragma unroll
    for (int ma = 0; ma < 2; ++ma) {
        #pragma unroll
        for (int n = 0; n < 4; ++n) {
            const int ncol = n * 16 + lm;
            #pragma unroll
            for (int r = 0; r < 8; ++r) {
                long mrow = m0 + ma * 16 + r + lhi * 8;
                out[mrow * 64 + ncol] = acc[ma][n][r];
            }
        }
    }
}

// ---------------- BN stats: per-channel partial sums (deterministic two-pass) ----------------
__global__ __launch_bounds__(256) void k_stats_partial(const float* __restrict__ hbuf,
                                                       float* __restrict__ psum, float* __restrict__ psq) {
    __shared__ float ss[256], sq[256];
    int t = threadIdx.x;
    float s = 0.f, q = 0.f;
    for (long i = (long)blockIdx.x * 256 + t; i < NELEM; i += 65536L) {  // channel = t&63 invariant
        float v = hbuf[i]; s += v; q += v * v;
    }
    ss[t] = s; sq[t] = q;
    __syncthreads();
    if (t < 64) {
        float S = ss[t] + ss[t + 64] + ss[t + 128] + ss[t + 192];
        float Q = sq[t] + sq[t + 64] + sq[t + 128] + sq[t + 192];
        psum[blockIdx.x * 64 + t] = S;
        psq [blockIdx.x * 64 + t] = Q;
    }
}

__global__ void k_stats_final(const float* __restrict__ psum, const float* __restrict__ psq,
                              float* __restrict__ stats) {
    int c = threadIdx.x;
    if (c >= 64) return;
    float S = 0.f, Q = 0.f;
    for (int p = 0; p < 256; ++p) { S += psum[p * 64 + c]; Q += psq[p * 64 + c]; }
    float mean = S / (float)NPIX;
    float var  = Q / (float)NPIX - mean * mean;
    stats[c]      = mean;
    stats[64 + c] = rsqrtf(var + 1e-5f);
}

// ---------------- BN + ReLU -> padded NHWC bf16 (input for conv2) ----------------
__global__ __launch_bounds__(256) void k_bnrelu(const float* __restrict__ hbuf, const float* __restrict__ stats,
                                                const float* __restrict__ g, const float* __restrict__ bb,
                                                bf16* __restrict__ hnp) {
    long i = (long)blockIdx.x * 256 + threadIdx.x;   // unpadded NHWC index
    int c = (int)(i & 63);
    long pix = i >> 6;
    int b  = (int)(pix >> 12);
    int hw = (int)(pix & 4095);
    int hh = hw >> 6, w = hw & 63;
    float y = (hbuf[i] - stats[c]) * stats[64 + c] * g[c] + bb[c];
    hnp[(((long)b * 66 + hh + 1) * 66 + (w + 1)) * 64 + c] = (bf16)(y > 0.f ? y : 0.f);
}

// ---------------- BN2 + residual + ReLU + gated accumulate (LDS-tiled transpose) ----------------
__global__ __launch_bounds__(256) void k_finalize(const float* __restrict__ h2, const float* __restrict__ stats,
                                                  const float* __restrict__ g, const float* __restrict__ bb,
                                                  const float* __restrict__ x, const float* __restrict__ gates,
                                                  int e, float* __restrict__ out) {
    __shared__ float tile[64][65];
    const int b   = blockIdx.x >> 6;
    const int row = blockIdx.x & 63;
    const float wgt = gates[b * 8 + e];
    if (wgt == 0.f) return;                           // uniform per block
    const int t = threadIdx.x;
    const float* src = h2 + ((long)b * 4096 + row * 64) * 64;   // + w*64 + c
    for (int i = t; i < 4096; i += 256) {             // coalesced: c inner
        int w = i >> 6, c = i & 63;
        float v = src[(long)w * 64 + c];
        tile[w][c] = (v - stats[c]) * stats[64 + c] * g[c] + bb[c];
    }
    __syncthreads();
    const long obase = (long)b * 64 * 4096 + row * 64;          // + c*4096 + w
    for (int i = t; i < 4096; i += 256) {             // coalesced: w inner
        int c = i >> 6, w = i & 63;
        long oi = obase + (long)c * 4096 + w;
        float r = tile[w][c] + x[oi];
        if (r < 0.f) r = 0.f;
        out[oi] += wgt * r;
    }
}

extern "C" void kernel_launch(void* const* d_in, const int* in_sizes, int n_in,
                              void* d_out, int out_size, void* d_ws, size_t ws_size,
                              hipStream_t stream) {
    const float* x   = (const float*)d_in[0];
    const float* rw  = (const float*)d_in[1];
    const float* c1w = (const float*)d_in[2];
    const float* g1  = (const float*)d_in[3];
    const float* b1  = (const float*)d_in[4];
    const float* c2w = (const float*)d_in[5];
    const float* g2  = (const float*)d_in[6];
    const float* b2  = (const float*)d_in[7];
    float* out = (float*)d_out;

    char* ws = (char*)d_ws;
    auto alloc = [&](size_t bytes) {
        char* p = ws;
        ws += (bytes + 255) & ~(size_t)255;
        return p;
    };
    float* gates  = (float*)alloc(256 * 4);
    float* pooled = (float*)alloc(2048 * 4);
    float* stats1 = (float*)alloc(128 * 4);
    float* stats2 = (float*)alloc(128 * 4);
    float* psum   = (float*)alloc(16384 * 4);
    float* psq    = (float*)alloc(16384 * 4);
    bf16*  wt1    = (bf16*)alloc(294912 * 2);
    bf16*  wt2    = (bf16*)alloc(294912 * 2);
    bf16*  xp     = (bf16*)alloc(PADELEM * 2);   // padded NHWC bf16 input
    bf16*  hnp    = (bf16*)alloc(PADELEM * 2);   // padded NHWC bf16 bn1-relu output
    float* hbuf   = (float*)alloc(NELEM * 4);    // conv output (reused conv1/conv2)

    const size_t convLds = 576 * LWSTRIDE * sizeof(bf16);   // 82944 B

    hipMemsetAsync(d_out, 0, (size_t)out_size * sizeof(float), stream);
    hipMemsetAsync(xp,  0, PADELEM * 2, stream);
    hipMemsetAsync(hnp, 0, PADELEM * 2, stream);

    k_x2nhwc<<<2048, 256, 0, stream>>>(x, xp);
    k_wprep <<<(2 * 294912 + 255) / 256, 256, 0, stream>>>(c1w, c2w, wt1, wt2);
    k_pool  <<<2048, 256, 0, stream>>>(x, pooled);
    k_router<<<1, 256, 0, stream>>>(pooled, rw, gates, out + NELEM);

    for (int e = 0; e < 8; ++e) {
        k_conv<<<512, 256, convLds, stream>>>(xp, wt1 + (size_t)e * 36864, hbuf);
        k_stats_partial<<<256, 256, 0, stream>>>(hbuf, psum, psq);
        k_stats_final<<<1, 64, 0, stream>>>(psum, psq, stats1);
        k_bnrelu<<<32768, 256, 0, stream>>>(hbuf, stats1, g1 + e * 64, b1 + e * 64, hnp);
        k_conv<<<512, 256, convLds, stream>>>(hnp, wt2 + (size_t)e * 36864, hbuf);
        k_stats_partial<<<256, 256, 0, stream>>>(hbuf, psum, psq);
        k_stats_final<<<1, 64, 0, stream>>>(psum, psq, stats2);
        k_finalize<<<2048, 256, 0, stream>>>(hbuf, stats2, g2 + e * 64, b2 + e * 64, x, gates, e, out);
    }
}
